// RecQNetwork_44375602103003
// MI455X (gfx1250) — matrix operand order repaired and checked by reading, not compiled, not run on hardware
//
#include <hip/hip_runtime.h>
#include <hip/hip_bf16.h>

// Sizes from the reference
#define TT  512
#define BB  128
#define AA  8
#define BA  1024          // B*A
#define OO  128
#define HH  128
#define H3  384
#define NAC 32
#define NROWS (TT*BA)     // 524288 rows of the batched GEMMs

typedef __attribute__((ext_vector_type(16))) _Float16 v16h;
typedef __attribute__((ext_vector_type(8)))  _Float16 v8h;
typedef __attribute__((ext_vector_type(4)))  _Float16 v4h;
typedef __attribute__((ext_vector_type(8)))  float    v8f;

// ---- packed-weight fragment layout -------------------------------------
// Every weight matrix here has K=128 -> KT=4 K-chunks of 32.
// frag id within a matrix = nt*4 + kt. Each frag = 32 lanes * 16 halfs = 512 halfs.
// Lane l holds column n = nt*16 + (l&15), K = kt*32 + ((l>=16)?8:0) + (j<8 ? j : j+8), j=0..15.
#define FRAG_HALFS 512
#define WPRE_FRAG0   0      // 8*4  = 32 frags
#define WI_FRAG0     32     // 24*4 = 96 frags
#define WH_FRAG0     128    // 24*4 = 96 frags
#define WPOST_FRAG0  224    // 8*4  = 32 frags
#define WOUT_FRAG0   256    // 2*4  = 8 frags
#define NFRAGS       264

// workspace layout (bytes):
//   [0, 264*512*2)                    packed f16 weights (~264KB)
//   [GI_OFF, GI_OFF + NROWS*384*4)    gi  fp32  (805 MB)
//   [YS_OFF, YS_OFF + NROWS*128*2)    ys  f16   (134 MB)
#define GI_OFF ((size_t)524288)
#define YS_OFF (GI_OFF + (size_t)NROWS*H3*4)

__device__ __forceinline__ v8f wmma32(v16h a, v16h b, v8f c) {
  return __builtin_amdgcn_wmma_f32_16x16x32_f16(false, a, false, b, (short)0, c,
                                                false, false);
}

// A fragment (16x32 f16) from a row-major 16x128 f16 LDS tile, K-chunk kc.
// ISA layout: lanes 0-15 -> M=lane, K=kc*32+{0..7,16..23}; lanes 16-31 -> K +8.
__device__ __forceinline__ v16h a_frag_from_lds(const _Float16* base, int kc, int lane) {
  int m  = lane & 15;
  int kb = kc * 32 + ((lane >> 4) << 3);
  const _Float16* p = base + m * HH + kb;
  v8h lo = *(const v8h*)p;          // K = kb .. kb+7
  v8h hi = *(const v8h*)(p + 16);   // K = kb+16 .. kb+23
  v16h a;
#pragma unroll
  for (int i = 0; i < 8; i++) { a[i] = lo[i]; a[i + 8] = hi[i]; }
  return a;
}

__device__ __forceinline__ v16h b_frag(const _Float16* pack, int fragid, int lane) {
  return *(const v16h*)(pack + (size_t)fragid * FRAG_HALFS + lane * 16);
}

// ---- kernel: pack all weights fp32 -> f16 B-fragment layout ------------
__global__ __launch_bounds__(256) void pack_weights(
    const float* __restrict__ Wpre, const float* __restrict__ Wi,
    const float* __restrict__ Wh,   const float* __restrict__ Wpost,
    const float* __restrict__ Wout, _Float16* __restrict__ dst) {
  int tid = blockIdx.x * 256 + threadIdx.x;
  if (tid >= NFRAGS * 32) return;
  int lane = tid & 31, frag = tid >> 5;
  const float* W; int N; int f;
  if      (frag < WI_FRAG0)    { W = Wpre;  N = HH;  f = frag; }
  else if (frag < WH_FRAG0)    { W = Wi;    N = H3;  f = frag - WI_FRAG0; }
  else if (frag < WPOST_FRAG0) { W = Wh;    N = H3;  f = frag - WH_FRAG0; }
  else if (frag < WOUT_FRAG0)  { W = Wpost; N = HH;  f = frag - WPOST_FRAG0; }
  else                         { W = Wout;  N = NAC; f = frag - WOUT_FRAG0; }
  int nt = f >> 2, kt = f & 3;
  int kb = kt * 32 + ((lane >> 4) << 3);
  int n  = nt * 16 + (lane & 15);
  _Float16* d = dst + (size_t)frag * FRAG_HALFS + lane * 16;
#pragma unroll
  for (int j = 0; j < 16; j++) d[j] = (_Float16)W[(size_t)(kb + (j < 8 ? j : j + 8)) * N + n];
}

// ---- kernel: gi = relu(obs @ W_pre + b_pre) @ Wi + bi ------------------
__global__ __launch_bounds__(256) void pre_gi_kernel(
    const float* __restrict__ obs, const _Float16* __restrict__ packW,
    const float* __restrict__ b_pre, const float* __restrict__ bi,
    float* __restrict__ gi) {
  __shared__ _Float16 sm[8][16 * HH];
  int lane = threadIdx.x & 31, w = threadIdx.x >> 5;
  _Float16* S = sm[w];
  size_t rowbase = ((size_t)blockIdx.x * 8 + w) * 16;   // 16-row tile
  int chi = (lane >> 4) << 3;   // C-layout M offset (lanes 16-31 -> +8)
  int cn  = lane & 15;          // C-layout column within N-tile

  // obs fp32 -> f16 LDS tile (each lane covers 4 columns per row)
#pragma unroll 4
  for (int r = 0; r < 16; r++) {
    float4 o4 = *(const float4*)(obs + (rowbase + r) * OO + lane * 4);
    v4h h4; h4[0] = (_Float16)o4.x; h4[1] = (_Float16)o4.y;
            h4[2] = (_Float16)o4.z; h4[3] = (_Float16)o4.w;
    *(v4h*)(S + r * HH + lane * 4) = h4;
  }
  __syncthreads();
  v16h A[4];
#pragma unroll
  for (int k = 0; k < 4; k++) A[k] = a_frag_from_lds(S, k, lane);

  // GEMM 1: emb = relu(obs @ W_pre + b_pre) -> back to LDS as f16
#pragma unroll
  for (int nt = 0; nt < 8; nt++) {
    v8f acc = {};
#pragma unroll
    for (int k = 0; k < 4; k++)
      acc = wmma32(A[k], b_frag(packW, WPRE_FRAG0 + nt * 4 + k, lane), acc);
    float bb = b_pre[nt * 16 + cn];
#pragma unroll
    for (int v = 0; v < 8; v++) {
      float e = acc[v] + bb;
      e = e > 0.f ? e : 0.f;
      S[(v + chi) * HH + nt * 16 + cn] = (_Float16)e;
    }
  }
  __syncthreads();
  v16h E[4];
#pragma unroll
  for (int k = 0; k < 4; k++) E[k] = a_frag_from_lds(S, k, lane);

  // GEMM 2: gi = emb @ Wi + bi  -> fp32 scratch [NROWS][384]
  float* grow = gi + rowbase * H3;
  for (int nt = 0; nt < 24; nt++) {
    v8f acc = {};
#pragma unroll
    for (int k = 0; k < 4; k++)
      acc = wmma32(E[k], b_frag(packW, WI_FRAG0 + nt * 4 + k, lane), acc);
    float bb = bi[nt * 16 + cn];
#pragma unroll
    for (int v = 0; v < 8; v++)
      grow[(size_t)(v + chi) * H3 + nt * 16 + cn] = acc[v] + bb;
  }
}

// ---- kernel: sequential GRU scan (rows independent -> 64 blocks) -------
// Block = 16 rows of B*A, 8 waves; wave w owns h-columns [16w,16w+16).
// Wh fragments live in VGPRs for the whole scan (3 gates x 4 K = 96 VGPRs).
__global__ __launch_bounds__(256) void scan_kernel(
    const float* __restrict__ h0, const unsigned char* __restrict__ resets,
    const float* __restrict__ gi, const _Float16* __restrict__ packW,
    const float* __restrict__ bhn, _Float16* __restrict__ ys,
    float* __restrict__ hfin) {
  __shared__ _Float16 hbuf[16 * HH];
  int lane = threadIdx.x & 31, w = threadIdx.x >> 5;
  int rowbase = blockIdx.x * 16;          // row tile in B*A space
  int chi = (lane >> 4) << 3;
  int cn  = lane & 15;
  int col = w * 16 + cn;                  // owned h column

  v16h whf[3][4];                         // resident Wh fragments
#pragma unroll
  for (int g = 0; g < 3; g++)
#pragma unroll
    for (int k = 0; k < 4; k++)
      whf[g][k] = b_frag(packW, WH_FRAG0 + (g * 8 + w) * 4 + k, lane);
  float bn = bhn[col];

  v8f hC;                                 // h state, C layout, fp32
#pragma unroll
  for (int v = 0; v < 8; v++)
    hC[v] = h0[(size_t)(rowbase + v + chi) * HH + col];

  for (int t = 0; t < TT; t++) {
    const unsigned char* rs = resets + (size_t)t * BA + rowbase;
#pragma unroll
    for (int v = 0; v < 8; v++)
      if (rs[v + chi]) hC[v] = 0.f;

    __syncthreads();                      // prior A-frag reads finished
#pragma unroll
    for (int v = 0; v < 8; v++)
      hbuf[(v + chi) * HH + col] = (_Float16)hC[v];
    __syncthreads();                      // h tile visible to all waves

    v16h A[4];
#pragma unroll
    for (int k = 0; k < 4; k++) A[k] = a_frag_from_lds(hbuf, k, lane);

    v8f ar = {}, az = {}, an = {};
#pragma unroll
    for (int k = 0; k < 4; k++) {
      ar = wmma32(A[k], whf[0][k], ar);
      az = wmma32(A[k], whf[1][k], az);
      an = wmma32(A[k], whf[2][k], an);
    }

    const float* grow = gi + ((size_t)t * BA + rowbase) * H3;
    if (t + 1 < TT)                       // pull next step's gi into cache
      __builtin_prefetch(grow + (size_t)BA * H3 + (size_t)chi * H3 + col, 0, 1);
#pragma unroll
    for (int v = 0; v < 8; v++) {
      int m = v + chi;
      float ir  = grow[(size_t)m * H3 + col];
      float iz  = grow[(size_t)m * H3 + 128 + col];
      float inn = grow[(size_t)m * H3 + 256 + col];
      float r_ = 1.f / (1.f + __expf(-(ir + ar[v])));
      float z_ = 1.f / (1.f + __expf(-(iz + az[v])));
      float n_ = tanhf(inn + r_ * (an[v] + bn));
      hC[v] = (1.f - z_) * n_ + z_ * hC[v];
      ys[((size_t)t * BA + rowbase + m) * HH + col] = (_Float16)hC[v];
    }
  }
#pragma unroll
  for (int v = 0; v < 8; v++)
    hfin[(size_t)(rowbase + v + chi) * HH + col] = hC[v];
}

// ---- kernel: q = relu(ys @ W_post + b_post) @ W_out + b_out ------------
__global__ __launch_bounds__(256) void post_out_kernel(
    const _Float16* __restrict__ ys, const _Float16* __restrict__ packW,
    const float* __restrict__ b_post, const float* __restrict__ b_out,
    float* __restrict__ q) {
  __shared__ _Float16 sm[8][16 * HH];
  int lane = threadIdx.x & 31, w = threadIdx.x >> 5;
  _Float16* S = sm[w];
  size_t rowbase = ((size_t)blockIdx.x * 8 + w) * 16;
  int chi = (lane >> 4) << 3;
  int cn  = lane & 15;

#pragma unroll 4
  for (int r = 0; r < 16; r++)
    *(v4h*)(S + r * HH + lane * 4) =
        *(const v4h*)(ys + (rowbase + r) * HH + lane * 4);
  __syncthreads();
  v16h A[4];
#pragma unroll
  for (int k = 0; k < 4; k++) A[k] = a_frag_from_lds(S, k, lane);

#pragma unroll
  for (int nt = 0; nt < 8; nt++) {
    v8f acc = {};
#pragma unroll
    for (int k = 0; k < 4; k++)
      acc = wmma32(A[k], b_frag(packW, WPOST_FRAG0 + nt * 4 + k, lane), acc);
    float bb = b_post[nt * 16 + cn];
#pragma unroll
    for (int v = 0; v < 8; v++) {
      float e = acc[v] + bb;
      e = e > 0.f ? e : 0.f;
      S[(v + chi) * HH + nt * 16 + cn] = (_Float16)e;
    }
  }
  __syncthreads();
  v16h P[4];
#pragma unroll
  for (int k = 0; k < 4; k++) P[k] = a_frag_from_lds(S, k, lane);

#pragma unroll
  for (int nt = 0; nt < 2; nt++) {
    v8f acc = {};
#pragma unroll
    for (int k = 0; k < 4; k++)
      acc = wmma32(P[k], b_frag(packW, WOUT_FRAG0 + nt * 4 + k, lane), acc);
    float bb = b_out[nt * 16 + cn];
#pragma unroll
    for (int v = 0; v < 8; v++)
      q[(rowbase + v + chi) * NAC + nt * 16 + cn] = acc[v] + bb;
  }
}

extern "C" void kernel_launch(void* const* d_in, const int* in_sizes, int n_in,
                              void* d_out, int out_size, void* d_ws, size_t ws_size,
                              hipStream_t stream) {
  (void)in_sizes; (void)n_in; (void)out_size; (void)ws_size;
  const float*         h0     = (const float*)d_in[0];
  const float*         obs    = (const float*)d_in[1];
  const unsigned char* resets = (const unsigned char*)d_in[2];  // jnp.bool_ = 1 byte
  const float*         W_pre  = (const float*)d_in[3];
  const float*         b_pre  = (const float*)d_in[4];
  const float*         Wi     = (const float*)d_in[5];
  const float*         bi     = (const float*)d_in[6];
  const float*         Wh     = (const float*)d_in[7];
  const float*         bhn    = (const float*)d_in[8];
  const float*         W_post = (const float*)d_in[9];
  const float*         b_post = (const float*)d_in[10];
  const float*         W_out  = (const float*)d_in[11];
  const float*         b_out  = (const float*)d_in[12];

  float* out = (float*)d_out;                    // [h_final(131072) | q(16777216)]
  _Float16* packW = (_Float16*)d_ws;
  float*    gi    = (float*)((char*)d_ws + GI_OFF);
  _Float16* ys    = (_Float16*)((char*)d_ws + YS_OFF);

  pack_weights<<<(NFRAGS * 32 + 255) / 256, 256, 0, stream>>>(
      W_pre, Wi, Wh, W_post, W_out, packW);

  // 524288 rows / 16 per wave / 8 waves per block = 4096 blocks
  pre_gi_kernel<<<4096, 256, 0, stream>>>(obs, packW, b_pre, bi, gi);

  scan_kernel<<<BA / 16, 256, 0, stream>>>(h0, resets, gi, packW, bhn, ys, out);

  post_out_kernel<<<4096, 256, 0, stream>>>(ys, packW, b_post, b_out,
                                            out + (size_t)BA * HH);
}
